// ResNet101RoIHead_74715251081430
// MI455X (gfx1250) — matrix operand order repaired and checked by reading, not compiled, not run on hardware
//
#include <hip/hip_runtime.h>
#include <hip/hip_bf16.h>
#include <math.h>

typedef __attribute__((ext_vector_type(16))) _Float16 v16h;
typedef __attribute__((ext_vector_type(8)))  _Float16 v8h;
typedef __attribute__((ext_vector_type(8)))  float    v8f;

union Frag16 { v16h v; v8h h[2]; };

#define H_     50
#define W_     50
#define C_IN   1024
#define C_OUT  2048
#define R_     256
#define OUT_   14
#define PADW   16                 // 14x14 interior + zero halo -> 16x16
#define KPOOL  6
#define NPIX   49                 // 7*7 conv-output pixels per ROI
#define NTOT   (R_ * NPIX)        // 12544
#define KDIM   (C_IN * 9)         // 9216
#define KTS    (KDIM / 32)        // 288 K-steps

// ---------------------------------------------------------------------------
// Kernel 1: ROI max-pool (fp32 in, f16 out) into zero-halo layout
//   pooled[r][c][16][16]; interior rows/cols 1..14 hold the 14x14 pool,
//   border is 0 so the conv gather needs no bounds checks.
// ---------------------------------------------------------------------------
__global__ __launch_bounds__(256) void roi_pool_f16_kernel(
    const float* __restrict__ fm, const float* __restrict__ rois,
    _Float16* __restrict__ pooled)
{
  int idx = blockIdx.x * 256 + threadIdx.x;       // [r][c][16][16]
  int px = idx & 15; int t = idx >> 4;
  int py = t & 15;   t >>= 4;
  int c  = t & (C_IN - 1); int r = t >> 10;

  int oy = py - 1, ox = px - 1;
  if ((unsigned)oy >= (unsigned)OUT_ || (unsigned)ox >= (unsigned)OUT_) {
    pooled[idx] = (_Float16)0.0f;                 // zero halo
    return;
  }

  float4 roi = reinterpret_cast<const float4*>(rois)[r];
  int xmin = (int)roundf(roi.x * 0.0625f);
  int ymin = (int)roundf(roi.y * 0.0625f);
  int xmax = (int)roundf(roi.z * 0.0625f);
  int ymax = (int)roundf(roi.w * 0.0625f);
  float roi_w = fmaxf((float)(xmax - xmin + 1), 1.0f);
  float roi_h = fmaxf((float)(ymax - ymin + 1), 1.0f);
  float sh = roi_h * (1.0f / OUT_), sw = roi_w * (1.0f / OUT_);
  int hs = min(max(ymin + (int)floorf((float)oy * sh), 0), H_);
  int he = min(max(ymin + (int)ceilf((float)(oy + 1) * sh), 0), H_);
  int ws_ = min(max(xmin + (int)floorf((float)ox * sw), 0), W_);
  int we  = min(max(xmin + (int)ceilf((float)(ox + 1) * sw), 0), W_);

  const float* fmc = fm + (size_t)c * (H_ * W_);
  float mx = -INFINITY;
  for (int ky = 0; ky < KPOOL; ++ky) {
    int h = hs + ky;
    if (h < he) {
      const float* rowp = fmc + min(h, H_ - 1) * W_;
      for (int kx = 0; kx < KPOOL; ++kx) {
        int w = ws_ + kx;
        if (w < we) mx = fmaxf(mx, rowp[min(w, W_ - 1)]);
      }
    }
  }
  pooled[idx] = (_Float16)mx;
}

// ---------------------------------------------------------------------------
// Kernel 2: conv_w (2048,1024,3,3) fp32 -> f16 in WMMA A-fragment order:
//   wA[k_tile][m_tile][lane][e], m = m_tile*16 + lane%16,
//   k = k_tile*32 + (e>=8?16:0) + (lane>=16?8:0) + (e&7)   [ISA 7.12.2]
// ---------------------------------------------------------------------------
__global__ __launch_bounds__(256) void weight_swizzle_kernel(
    const float* __restrict__ conv_w, _Float16* __restrict__ wA)
{
  int idx = blockIdx.x * 256 + threadIdx.x;           // half index, < 18874368
  int e      = idx & 15;
  int lane   = (idx >> 4) & 31;
  int m_tile = (idx >> 9) & 127;
  int k_tile = idx >> 16;
  int m = m_tile * 16 + (lane & 15);
  int k = k_tile * 32 + ((e >= 8) ? 16 : 0) + ((lane >= 16) ? 8 : 0) + (e & 7);
  wA[idx] = (_Float16)conv_w[(size_t)m * KDIM + k];   // k == c*9 + ky*3 + kx
}

// ---------------------------------------------------------------------------
// Kernel 3: implicit-GEMM conv via WMMA.
//   Block tile 256(M) x 64(N), 8 waves = 4 m-slabs (64 rows) x 2 n-slabs (32).
//   Each wave: 4 A-frags x 2 B-frags = 8 wmma per 32-wide K step.
//   A fragments direct from pre-swizzled global (L2-resident, prefetched);
//   B (im2col over zero-halo pooled) gathered branch-free into LDS once per
//   256-row M block (half the gather traffic of a 128-row block).
//   Epilogue: relu(acc + bias) -> f16 xbuf[n][m].
// ---------------------------------------------------------------------------
__global__ __launch_bounds__(256) void conv_gemm_wmma_kernel(
    const _Float16* __restrict__ wA,
    const _Float16* __restrict__ pooled,
    const float* __restrict__ conv_b,
    _Float16* __restrict__ xbuf)
{
  __shared__ __align__(16) _Float16 lds_b[32][72];   // 144B row stride: 16B-aligned

  const int t = threadIdx.x;
  const int lane = t & 31;
  const int wid  = t >> 5;
  const int wave_m = wid >> 1;                  // 0..3 -> 64-row slab
  const int wave_n = wid & 1;                   // 0..1 -> 32-col slab
  const int n0 = blockIdx.x * 64;
  const int mtg = blockIdx.y * 16 + wave_m * 4; // global 16-row tile index

  // B-gather mapping: thread -> (k row, 8 consecutive n)
  const int kb = t >> 3;                        // 0..31
  const int nb = (t & 7) * 8;                   // 0..56

  // per-n constants: base half-offset into halo pooled for (roi, oy*2, ox*2)
  int base_n[8];
#pragma unroll
  for (int j = 0; j < 8; ++j) {
    int n   = n0 + nb + j;
    int roi = n / NPIX;
    int pix = n - roi * NPIX;
    int oy  = pix / 7;
    int ox  = pix - oy * 7;
    base_n[j] = roi * (C_IN * PADW * PADW) + (oy * 2) * PADW + (ox * 2);
  }

  // incremental k decode: k = kt*32 + kb = cch*9 + rem
  int cch = kb / 9;
  int rem = kb - cch * 9;

  v8f acc[4][2];
#pragma unroll
  for (int mi = 0; mi < 4; ++mi)
#pragma unroll
    for (int nf = 0; nf < 2; ++nf) acc[mi][nf] = (v8f){};

  const size_t lane_off = (size_t)lane * 16;    // 16 halves = 32B per lane

  for (int kt = 0; kt < KTS; ++kt) {
    // ---- branch-free im2col gather of B tile (32 x 64) ----
    // ky*16+kx == rem + 13*(rem/3), rem/3 == (rem*11)>>5 for rem in [0,8]
    int off_k = cch * (PADW * PADW) + rem + 13 * ((rem * 11) >> 5);
    v8h pack;
#pragma unroll
    for (int j = 0; j < 8; ++j)
      pack[j] = pooled[base_n[j] + off_k];
    // advance k by 32 = 3*9 + 5
    rem += 5; cch += 3;
    if (rem >= 9) { rem -= 9; cch += 1; }

    __syncthreads();
    *reinterpret_cast<v8h*>(&lds_b[kb][nb]) = pack;
    __syncthreads();

    // ---- A fragments: coalesced 32B/lane from pre-swizzled global ----
    const _Float16* base = wA + ((size_t)kt * 128 + mtg) * 512;
    Frag16 a[4];
#pragma unroll
    for (int mi = 0; mi < 4; ++mi) {
      const v8h* pa = reinterpret_cast<const v8h*>(base + (size_t)mi * 512 + lane_off);
      a[mi].h[0] = pa[0];
      a[mi].h[1] = pa[1];
    }
    if (kt + 1 < KTS) {
      __builtin_prefetch((const void*)(base + 128 * 512 + lane_off), 0, 1);
      __builtin_prefetch((const void*)(base + 128 * 512 + 1024 + lane_off), 0, 1);
    }

    // ---- B fragments from LDS (n = lane%16, k = table order over elements) ----
    const int nbase = wave_n * 32;
    Frag16 b0, b1;
    const int nl = lane & 15;
    const int kh = (lane >> 4) * 8;
#pragma unroll
    for (int s = 0; s < 8; ++s) {
      b0.v[s]     = lds_b[kh + s][nbase + nl];
      b0.v[8 + s] = lds_b[16 + kh + s][nbase + nl];
      b1.v[s]     = lds_b[kh + s][nbase + 16 + nl];
      b1.v[8 + s] = lds_b[16 + kh + s][nbase + 16 + nl];
    }

#pragma unroll
    for (int mi = 0; mi < 4; ++mi) {
      acc[mi][0] = __builtin_amdgcn_wmma_f32_16x16x32_f16(
          false, a[mi].v, false, b0.v, (short)0, acc[mi][0], false, false);
      acc[mi][1] = __builtin_amdgcn_wmma_f32_16x16x32_f16(
          false, a[mi].v, false, b1.v, (short)0, acc[mi][1], false, false);
    }
  }

  // ---- epilogue: bias + relu, f16 store (C layout: m=i+(lane>=16?8:0), n=lane%16)
  const int n_lane = lane & 15;
  const int m_half = (lane >= 16) ? 8 : 0;
#pragma unroll
  for (int mi = 0; mi < 4; ++mi) {
    int m_start = blockIdx.y * 256 + wave_m * 64 + mi * 16 + m_half;
#pragma unroll
    for (int nf = 0; nf < 2; ++nf) {
      int n = n0 + wave_n * 32 + nf * 16 + n_lane;
      v8h o;
#pragma unroll
      for (int i = 0; i < 8; ++i) {
        float v = acc[mi][nf][i] + conv_b[m_start + i];
        o[i] = (_Float16)fmaxf(v, 0.0f);
      }
      *reinterpret_cast<v8h*>(xbuf + (size_t)n * C_OUT + m_start) = o;
    }
  }
}

// ---------------------------------------------------------------------------
// Kernel 4: fc7[r][m] = mean over 49 pixels of xbuf[n][m]   (deterministic)
// ---------------------------------------------------------------------------
__global__ __launch_bounds__(256) void reduce_fc7_kernel(
    const _Float16* __restrict__ xbuf, float* __restrict__ fc7)
{
  int idx = blockIdx.x * 256 + threadIdx.x;   // < 256*2048
  int m = idx & (C_OUT - 1);
  int r = idx >> 11;
  const _Float16* p = xbuf + (size_t)r * NPIX * C_OUT + m;
  float s = 0.0f;
#pragma unroll 7
  for (int i = 0; i < NPIX; ++i) s += (float)p[(size_t)i * C_OUT];
  fc7[idx] = s * (1.0f / 49.0f);
}

// ---------------------------------------------------------------------------
// Kernel 5: per-ROI FC heads + softmax. out = [cls_locs(256x84) | scores(256x21)]
// ---------------------------------------------------------------------------
__global__ __launch_bounds__(128) void fc_head_kernel(
    const float* __restrict__ fc7,
    const float* __restrict__ clw, const float* __restrict__ clb,
    const float* __restrict__ scw, const float* __restrict__ scb,
    float* __restrict__ out)
{
  __shared__ float row[C_OUT];
  __shared__ float sc[21];
  const int r = blockIdx.x, t = threadIdx.x;
  for (int i = t; i < C_OUT; i += 128) row[i] = fc7[(size_t)r * C_OUT + i];
  __syncthreads();

  if (t < 84) {
    const float* w = clw + (size_t)t * C_OUT;
    float acc = clb[t];
    for (int k2 = 0; k2 < C_OUT; ++k2) acc = fmaf(row[k2], w[k2], acc);
    out[(size_t)r * 84 + t] = acc;
  } else if (t < 105) {
    int j = t - 84;
    const float* w = scw + (size_t)j * C_OUT;
    float acc = scb[j];
    for (int k2 = 0; k2 < C_OUT; ++k2) acc = fmaf(row[k2], w[k2], acc);
    sc[j] = acc;
  }
  __syncthreads();
  if (t < 21) {
    float mx = -INFINITY;
    for (int j = 0; j < 21; ++j) mx = fmaxf(mx, sc[j]);
    float sum = 0.0f;
    for (int j = 0; j < 21; ++j) sum += expf(sc[j] - mx);
    out[(size_t)R_ * 84 + (size_t)r * 21 + t] = expf(sc[t] - mx) / sum;
  }
}

// ---------------------------------------------------------------------------
extern "C" void kernel_launch(void* const* d_in, const int* in_sizes, int n_in,
                              void* d_out, int out_size, void* d_ws, size_t ws_size,
                              hipStream_t stream)
{
  const float* fm     = (const float*)d_in[0];
  const float* rois   = (const float*)d_in[1];
  // d_in[2] = roi_indices: batch size 1, all zeros -> unused
  const float* conv_w = (const float*)d_in[3];
  const float* conv_b = (const float*)d_in[4];
  const float* clw    = (const float*)d_in[5];
  const float* clb    = (const float*)d_in[6];
  const float* scw    = (const float*)d_in[7];
  const float* scb    = (const float*)d_in[8];
  float* out = (float*)d_out;

  char* ws = (char*)d_ws;
  _Float16* pooled = (_Float16*)(ws);                 // 134,217,728 B (halo 16x16)
  _Float16* wA     = (_Float16*)(ws + 134217728);     //  37,748,736 B
  _Float16* xbuf   = (_Float16*)(ws + 171966464);     //  51,380,224 B
  float*    fc7    = (float*)   (ws + 223346688);     //   2,097,152 B

  roi_pool_f16_kernel<<<(R_ * C_IN * PADW * PADW) / 256, 256, 0, stream>>>(fm, rois, pooled);
  weight_swizzle_kernel<<<(KDIM * C_OUT) / 256, 256, 0, stream>>>(conv_w, wA);
  conv_gemm_wmma_kernel<<<dim3(NTOT / 64, C_OUT / 256), 256, 0, stream>>>(wA, pooled, conv_b, xbuf);
  reduce_fc7_kernel<<<(R_ * C_OUT) / 256, 256, 0, stream>>>(xbuf, fc7);
  fc_head_kernel<<<R_, 128, 0, stream>>>(fc7, clw, clb, scw, scb, out);
}
